// MultiHeadAttentionLinear_75694503624759
// MI455X (gfx1250) — matrix operand order repaired and checked by reading, not compiled, not run on hardware
//
#include <hip/hip_runtime.h>

#define BB 8
#define KT 4096
#define IND 512
#define HH 8
#define DD 64

typedef __attribute__((ext_vector_type(16))) __bf16 v16bf;
typedef __attribute__((ext_vector_type(8)))  __bf16 v8bf;
typedef __attribute__((ext_vector_type(8)))  float  v8f;

__device__ __forceinline__ __bf16 f2bf(float f) {
  union { float f; unsigned u; } v; v.f = f;
  unsigned r = v.u + 0x7FFFu + ((v.u >> 16) & 1u);   // round-to-nearest-even
  union { unsigned short s; __bf16 b; } o; o.s = (unsigned short)(r >> 16);
  return o.b;
}

__device__ __forceinline__ v16bf cat8(v8bf lo, v8bf hi) {
  return __builtin_shufflevector(lo, hi, 0,1,2,3,4,5,6,7,8,9,10,11,12,13,14,15);
}

__device__ __forceinline__ v8f wmma_bf16(v16bf a, v16bf b, v8f c) {
  return __builtin_amdgcn_wmma_f32_16x16x32_bf16(false, a, false, b, (short)0, c,
                                                 false, false);
}

// A fragment (16x32 bf16, row-major tile, K contiguous). ISA 16-bit A layout:
// lane half h: VGPR0-3 hold K = 8h..8h+7, VGPR4-7 hold K = 16+8h..16+8h+7.
__device__ __forceinline__ v16bf load_a(const __bf16* tile, int stride, int m, int half) {
  const __bf16* p = tile + (size_t)m * stride + 8 * half;
  return cat8(*(const v8bf*)p, *(const v8bf*)(p + 16));
}
// B fragment (32x16 bf16) from n-major storage (per-lane column n, K contiguous):
// lanes 0-15 K=0..15, lanes 16-31 K=16..31, packed 2-per-VGPR sequentially.
__device__ __forceinline__ v16bf load_b(const __bf16* tileT, int stride, int n, int half) {
  const __bf16* p = tileT + (size_t)n * stride + 16 * half;
  return cat8(*(const v8bf*)p, *(const v8bf*)(p + 8));
}

// ---------------- kernel 0: fp32 weights -> bf16, transposed (n-major, k contiguous)
__global__ __launch_bounds__(256) void prep_weights(
    const float* __restrict__ Wq, const float* __restrict__ Wk,
    const float* __restrict__ Wv, const float* __restrict__ Wp,
    __bf16* __restrict__ WqkT, __bf16* __restrict__ WvpT) {
  int idx = blockIdx.x * 256 + threadIdx.x;
  if (idx < 1024 * 512) {
    int n = idx >> 9, k = idx & 511;
    float w = (n < 512) ? Wq[(size_t)k * 512 + n] : Wk[(size_t)k * 512 + (n - 512)];
    WqkT[idx] = f2bf(w);
  } else if (idx < 1024 * 512 + 576 * 512) {
    int j = idx - 1024 * 512;
    int n = j >> 9, k = j & 511;
    float w = (n < 512) ? Wv[(size_t)k * 512 + n] : Wp[(size_t)k * 64 + (n - 512)];
    WvpT[j] = f2bf(w);
  }
}

// ---------------- kernel A: fused QKV + yp projections
// path 0: x @ [Wq|Wk] -> Qt,Kt stored transposed (h,b,d,k) bf16 (packed b128 stores)
// path 1: y @ [Wv|Wp] computed TRANSPOSED (A=W^T, B=x-tile) so each lane owns one
//         token and 8 contiguous output cols -> packed stores for V (h,b,k,e) + yp.
__global__ __launch_bounds__(256) void proj_kernel(
    const float* __restrict__ X, const float* __restrict__ Y,
    const float* __restrict__ bq, const float* __restrict__ bk,
    const float* __restrict__ bv,
    const __bf16* __restrict__ WqkT, const __bf16* __restrict__ WvpT,
    __bf16* __restrict__ Qt, __bf16* __restrict__ Kt,
    __bf16* __restrict__ V, float* __restrict__ yp) {
  __shared__ __attribute__((aligned(16))) __bf16 sx[32 * 520];
  const int k0 = blockIdx.x * 32;
  const int b = blockIdx.y;
  const int path = blockIdx.z;
  const float* src = ((path == 0) ? X : Y) + ((size_t)b * KT + k0) * IND;

  for (int i = threadIdx.x; i < 32 * 512; i += 256) {
    int r = i >> 9, c = i & 511;
    sx[r * 520 + c] = f2bf(src[(size_t)r * IND + c]);
  }
  __syncthreads();

  const int w = threadIdx.x >> 5;
  const int lane = threadIdx.x & 31;
  const int li = lane & 15, half = lane >> 4;

  if (path == 0) {
    // C[token, col]: A = x tile (LDS), B = WqkT (global, n-major)
    for (int nt = w; nt < 64; nt += 8) {
      v8f acc0 = {0.f, 0.f, 0.f, 0.f, 0.f, 0.f, 0.f, 0.f};
      v8f acc1 = acc0;
      const __bf16* wrow = WqkT + ((size_t)nt * 16 + li) * 512;
      __builtin_prefetch(wrow + 16 * 512, 0, 0);
#pragma unroll 4
      for (int ks = 0; ks < 16; ++ks) {
        const __bf16* p = wrow + ks * 32 + 8 * half;
        v16bf bf = cat8(*(const v8bf*)p, *(const v8bf*)(p + 16));
        v16bf a0 = load_a(&sx[ks * 32], 520, li, half);
        acc0 = wmma_bf16(a0, bf, acc0);
        v16bf a1 = load_a(&sx[16 * 520 + ks * 32], 520, li, half);
        acc1 = wmma_bf16(a1, bf, acc1);
      }
      const int n = nt * 16 + li;
      float bias = (n < 512) ? bq[n] : bk[n - 512];
      int hh = (n & 511) >> 6, d = n & 63;
      __bf16* dst = ((n < 512) ? Qt : Kt) + (((size_t)hh * BB + b) * 64 + d) * KT;
      {
        union { __bf16 bh[8]; uint4 u; } pk;
#pragma unroll
        for (int r = 0; r < 8; ++r) pk.bh[r] = f2bf(acc0[r] + bias);
        *(uint4*)(dst + k0 + 8 * half) = pk.u;
      }
      {
        union { __bf16 bh[8]; uint4 u; } pk;
#pragma unroll
        for (int r = 0; r < 8; ++r) pk.bh[r] = f2bf(acc1[r] + bias);
        *(uint4*)(dst + k0 + 16 + 8 * half) = pk.u;
      }
    }
  } else {
    // C^T[col, token]: A = WvpT rows (global), B = x tile rows (LDS, token-major)
    for (int mt = w; mt < 36; mt += 8) {
      v8f acc0 = {0.f, 0.f, 0.f, 0.f, 0.f, 0.f, 0.f, 0.f};  // token tile 0
      v8f acc1 = acc0;                                        // token tile 1
      const __bf16* wrow = WvpT + ((size_t)mt * 16 + li) * 512;
      __builtin_prefetch(wrow + 16 * 512, 0, 0);
#pragma unroll 4
      for (int ks = 0; ks < 16; ++ks) {
        const __bf16* pa = wrow + ks * 32 + 16 * half;
        v16bf af = cat8(*(const v8bf*)pa, *(const v8bf*)(pa + 8));
        const __bf16* pb0 = &sx[li * 520 + ks * 32 + 16 * half];
        v16bf b0 = cat8(*(const v8bf*)pb0, *(const v8bf*)(pb0 + 8));
        acc0 = wmma_bf16(af, b0, acc0);
        const __bf16* pb1 = &sx[(16 + li) * 520 + ks * 32 + 16 * half];
        v16bf b1 = cat8(*(const v8bf*)pb1, *(const v8bf*)(pb1 + 8));
        acc1 = wmma_bf16(af, b1, acc1);
      }
      const int col0 = mt * 16 + 8 * half;  // 8 contiguous output cols per lane
      const int ka = k0 + li, kb2 = k0 + 16 + li;
      if (col0 < 512) {
        int hh = mt >> 2, e0 = col0 & 63;
        __bf16* dstb = V + ((size_t)hh * BB + b) * (size_t)KT * 64;
        union { __bf16 bh[8]; uint4 u; } p0, p1;
#pragma unroll
        for (int r = 0; r < 8; ++r) {
          float bias = bv[col0 + r];
          p0.bh[r] = f2bf(acc0[r] + bias);
          p1.bh[r] = f2bf(acc1[r] + bias);
        }
        *(uint4*)(dstb + (size_t)ka * 64 + e0) = p0.u;
        *(uint4*)(dstb + (size_t)kb2 * 64 + e0) = p1.u;
      } else {
        int e0 = col0 - 512;
        float* dsty = yp + (size_t)b * KT * 64;
        float4 lo0 = {acc0[0], acc0[1], acc0[2], acc0[3]};
        float4 hi0 = {acc0[4], acc0[5], acc0[6], acc0[7]};
        float4 lo1 = {acc1[0], acc1[1], acc1[2], acc1[3]};
        float4 hi1 = {acc1[4], acc1[5], acc1[6], acc1[7]};
        *(float4*)(dsty + (size_t)ka * 64 + e0) = lo0;
        *(float4*)(dsty + (size_t)ka * 64 + e0 + 4) = hi0;
        *(float4*)(dsty + (size_t)kb2 * 64 + e0) = lo1;
        *(float4*)(dsty + (size_t)kb2 * 64 + e0 + 4) = hi1;
      }
    }
  }
}

// ---------------- kernel B: energy (q^T k over 4096 tokens) + softmax -> attn bf16
__global__ __launch_bounds__(256) void energy_softmax_kernel(
    const __bf16* __restrict__ Qt, const __bf16* __restrict__ Kt,
    __bf16* __restrict__ attnW) {
  __shared__ float E[64][65];
  const int h = blockIdx.x, b = blockIdx.y;
  const __bf16* qb = Qt + ((size_t)h * BB + b) * 64 * KT;
  const __bf16* kb = Kt + ((size_t)h * BB + b) * 64 * KT;
  const int w = threadIdx.x >> 5, lane = threadIdx.x & 31;
  const int li = lane & 15, half = lane >> 4;

  for (int t2 = 2 * w; t2 < 2 * w + 2; ++t2) {
    const int dt = t2 >> 2, et = t2 & 3;
    v8f acc = {0.f, 0.f, 0.f, 0.f, 0.f, 0.f, 0.f, 0.f};
    const __bf16* arow = qb + (size_t)(dt * 16 + li) * KT + 8 * half;
    const __bf16* brow = kb + (size_t)(et * 16 + li) * KT + 8 * half;
    for (int ks = 0; ks < 128; ++ks) {
      if ((ks & 15) == 0) {
        __builtin_prefetch(arow + ks * 32 + 1024, 0, 0);
        __builtin_prefetch(brow + ks * 32 + 1024, 0, 0);
      }
      v16bf a = cat8(*(const v8bf*)(arow + ks * 32), *(const v8bf*)(arow + ks * 32 + 16));
      v16bf bf = cat8(*(const v8bf*)(brow + ks * 32), *(const v8bf*)(brow + ks * 32 + 16));
      acc = wmma_bf16(a, bf, acc);
    }
#pragma unroll
    for (int r = 0; r < 8; ++r)
      E[dt * 16 + 8 * half + r][et * 16 + li] = acc[r];
  }
  __syncthreads();

  const int t = threadIdx.x;
  if (t < 64) {  // one thread per energy row; softmax over e
    float mx = -1e30f;
    for (int e = 0; e < 64; ++e) mx = fmaxf(mx, E[t][e]);
    float s = 0.f;
    for (int e = 0; e < 64; ++e) s += __expf(E[t][e] - mx);
    float inv = 1.f / s;
    __bf16* dst = attnW + ((size_t)(h * BB + b) * 64 + t) * 64;
    for (int e = 0; e < 64; ++e) dst[e] = f2bf(__expf(E[t][e] - mx) * inv);
  }
}

// ---------------- kernel C: out = g0 * (V @ attn^T) + g1 * yp  (coalesced epilogue)
__global__ __launch_bounds__(256) void out_kernel(
    const __bf16* __restrict__ V, const __bf16* __restrict__ attnW,
    const float* __restrict__ yp, const float* __restrict__ gamma,
    float* __restrict__ out) {
  __shared__ __attribute__((aligned(16))) __bf16 sattn[64 * 72];
  __shared__ float sE[64][65];
  const int k0 = blockIdx.x * 64;
  const int h = blockIdx.y, b = blockIdx.z;
  const __bf16* asrc = attnW + (size_t)(h * BB + b) * 64 * 64;

  // Async-copy attn tile (bf16, verbatim) global -> LDS with padded rows.
  // 512 x 16B chunks; per-lane LDS dest via VDST (ISA 10.x async GV mode).
  for (int c = threadIdx.x; c < 512; c += 256) {
    int q = c >> 3, e0 = (c & 7) * 8;
    unsigned lds_off = (unsigned)(size_t)&sattn[q * 72 + e0];
    const __bf16* g = asrc + q * 64 + e0;
    asm volatile("global_load_async_to_lds_b128 %0, %1, off"
                 :: "v"(lds_off), "v"(g) : "memory");
  }
  asm volatile("s_wait_asynccnt 0x0" ::: "memory");
  __syncthreads();

  const int w = threadIdx.x >> 5, lane = threadIdx.x & 31;
  const int li = lane & 15, half = lane >> 4;
  const __bf16* vb = V + ((size_t)h * BB + b) * (size_t)KT * 64;

  for (int t2 = 2 * w; t2 < 2 * w + 2; ++t2) {
    const int mt = t2 >> 2, nt = t2 & 3;
    v8f acc = {0.f, 0.f, 0.f, 0.f, 0.f, 0.f, 0.f, 0.f};
    const __bf16* arow = vb + (size_t)(k0 + mt * 16 + li) * 64 + 8 * half;
    const __bf16* brow = &sattn[(nt * 16 + li) * 72 + 8 * half];
#pragma unroll
    for (int ks = 0; ks < 2; ++ks) {
      v16bf a = cat8(*(const v8bf*)(arow + ks * 32), *(const v8bf*)(arow + ks * 32 + 16));
      v16bf bf = cat8(*(const v8bf*)(brow + ks * 32), *(const v8bf*)(brow + ks * 32 + 16));
      acc = wmma_bf16(a, bf, acc);
    }
#pragma unroll
    for (int r = 0; r < 8; ++r)
      sE[mt * 16 + 8 * half + r][nt * 16 + li] = acc[r];
  }
  __syncthreads();

  // Coalesced fused epilogue: float4 reads of yp, float4 writes of out.
  const float g = gamma[h];
  const float g0 = g / (1.f + g), g1 = 1.f / (1.f + g);
  for (int i = threadIdx.x; i < 64 * 16; i += 256) {
    int row = i >> 4, q0 = (i & 15) * 4;
    int kk = k0 + row;
    const float4 yv = *(const float4*)&yp[((size_t)b * KT + kk) * 64 + q0];
    float4 o;
    o.x = g0 * sE[row][q0 + 0] + g1 * yv.x;
    o.y = g0 * sE[row][q0 + 1] + g1 * yv.y;
    o.z = g0 * sE[row][q0 + 2] + g1 * yv.z;
    o.w = g0 * sE[row][q0 + 3] + g1 * yv.w;
    *(float4*)&out[(((size_t)b * HH + h) * KT + kk) * 64 + q0] = o;
  }
}

extern "C" void kernel_launch(void* const* d_in, const int* in_sizes, int n_in,
                              void* d_out, int out_size, void* d_ws, size_t ws_size,
                              hipStream_t stream) {
  (void)in_sizes; (void)n_in; (void)out_size; (void)ws_size;
  const float* x  = (const float*)d_in[0];
  const float* y  = (const float*)d_in[1];
  const float* Wq = (const float*)d_in[2];
  const float* bq = (const float*)d_in[3];
  const float* Wk = (const float*)d_in[4];
  const float* bk = (const float*)d_in[5];
  const float* Wv = (const float*)d_in[6];
  const float* bv = (const float*)d_in[7];
  const float* Wp = (const float*)d_in[8];
  const float* gm = (const float*)d_in[9];
  float* out = (float*)d_out;

  char* ws = (char*)d_ws;
  size_t off = 0;
  auto alloc = [&](size_t bytes) -> void* {
    void* p = ws + off;
    off = (off + bytes + 255) & ~(size_t)255;
    return p;
  };
  __bf16* WqkT = (__bf16*)alloc((size_t)1024 * 512 * 2);
  __bf16* WvpT = (__bf16*)alloc((size_t)576 * 512 * 2);
  __bf16* Qt   = (__bf16*)alloc((size_t)HH * BB * 64 * KT * 2);
  __bf16* Kt   = (__bf16*)alloc((size_t)HH * BB * 64 * KT * 2);
  __bf16* V    = (__bf16*)alloc((size_t)HH * BB * (size_t)KT * 64 * 2);
  float*  yp   = (float*)alloc((size_t)BB * KT * 64 * 4);
  __bf16* attn = (__bf16*)alloc((size_t)HH * BB * 64 * 64 * 2);

  prep_weights<<<dim3((1024 * 512 + 576 * 512 + 255) / 256), 256, 0, stream>>>(
      Wq, Wk, Wv, Wp, WqkT, WvpT);
  proj_kernel<<<dim3(KT / 32, BB, 2), 256, 0, stream>>>(
      x, y, bq, bk, bv, WqkT, WvpT, Qt, Kt, V, yp);
  energy_softmax_kernel<<<dim3(HH, BB), 256, 0, stream>>>(Qt, Kt, attn);
  out_kernel<<<dim3(KT / 64, HH, BB), 256, 0, stream>>>(V, attn, yp, gm, out);
}
